// InterpretableMultiHeadAttention_55009941127404
// MI455X (gfx1250) — compile-verified, matrix operand order checked by reading, not run on hardware
//
#include <hip/hip_runtime.h>
#include <hip/hip_bf16.h>

// ---------------- types ----------------
typedef __attribute__((ext_vector_type(16))) __bf16        v16bf;
typedef __attribute__((ext_vector_type(8)))  float         v8f;
typedef __attribute__((ext_vector_type(4)))  unsigned int  u32x4;
typedef int v4i_g __attribute__((vector_size(4 * sizeof(int))));  // GCC-style, matches builtin

#define B_ 8
#define S_ 1024
#define D_ 1024
#define H_ 16
#define DK_ 64

#if __has_builtin(__builtin_amdgcn_global_load_async_to_lds_b128) && \
    __has_builtin(__builtin_amdgcn_s_wait_asynccnt)
#define ASYNC_LDS 1
#else
#define ASYNC_LDS 0
#endif

#define AS1 __attribute__((address_space(1)))
#define AS3 __attribute__((address_space(3)))

// ---------------- helpers ----------------
__device__ __forceinline__ unsigned short f32_to_bf16(float f) {
  union { float f; unsigned int u; } v; v.f = f;
  unsigned int u = v.u;
  u += 0x7FFFu + ((u >> 16) & 1u);   // round-to-nearest-even
  return (unsigned short)(u >> 16);
}

union FragU { v16bf v; u32x4 q[2]; };

// Load a 16x32 bf16 WMMA fragment from a row-major matrix (16 rows, ld elems).
// Per ISA A-layout: lanes 0-15 (row=lane) hold K 0..7 & 16..23; lanes 16-31
// hold K 8..15 & 24..31.  Two 16B loads per lane.  Same pattern serves the
// B operand when the B matrix is stored pre-transposed as [N,K].
// Works for both global and LDS-resident tiles (ds_load_b128 from __shared__).
__device__ __forceinline__ v16bf load_frag(const unsigned short* __restrict__ p, int ld) {
  const int lane = threadIdx.x & 31;
  const int r    = lane & 15;
  const int hi   = lane >> 4;
  const unsigned short* base = p + (long)r * ld + hi * 8;
  FragU f;
  f.q[0] = *(const u32x4*)(base);
  f.q[1] = *(const u32x4*)(base + 16);
  return f.v;
}

__device__ __forceinline__ v8f wmma_bf16(v16bf a, v16bf b, v8f c) {
  return __builtin_amdgcn_wmma_f32_16x16x32_bf16(false, a, false, b, (short)0, c,
                                                 false, false);
}

#if ASYNC_LDS
// one 16-byte async copy global -> LDS (ASYNCcnt-tracked)
__device__ __forceinline__ void async_cp16(const unsigned short* g, unsigned short* l) {
  __builtin_amdgcn_global_load_async_to_lds_b128(
      (AS1 v4i_g*)(g), (AS3 v4i_g*)(l), 0, 0);
}
#endif

// ---------------- packing kernels ----------------
__global__ __launch_bounds__(256) void cast_f32_to_bf16(const float* __restrict__ in,
                                                        unsigned short* __restrict__ out,
                                                        long n) {
  long i = (long)blockIdx.x * 256 + threadIdx.x;
  if (i < n) out[i] = f32_to_bf16(in[i]);
}

// W_q/W_k [H,D,dk] -> Bt[n*D + d] = W[h,d,j], n = h*dk + j   (i.e. [N=H*dk, K=D])
__global__ __launch_bounds__(256) void pack_wqk(const float* __restrict__ W,
                                                unsigned short* __restrict__ Bt) {
  long i = (long)blockIdx.x * 256 + threadIdx.x;   // over D*D
  if (i >= (long)D_ * D_) return;
  int n = (int)(i / D_), d = (int)(i % D_);
  int h = n / DK_, j = n % DK_;
  Bt[i] = f32_to_bf16(W[((long)h * D_ + d) * DK_ + j]);
}

// W [K=D, N=D] -> Bt[n*D + d] = W[d*D + n]    (transpose to [N,K])
__global__ __launch_bounds__(256) void pack_wt(const float* __restrict__ W,
                                               unsigned short* __restrict__ Bt) {
  long i = (long)blockIdx.x * 256 + threadIdx.x;
  if (i >= (long)D_ * D_) return;
  int n = (int)(i / D_), d = (int)(i % D_);
  Bt[i] = f32_to_bf16(W[(long)d * D_ + n]);
}

// ---------------- generic batched WMMA GEMM ----------------
// C[m,n] = scale * sum_k A[m,k] * Bt[n,k]     (Bt is the [N,K] packed operand)
// 256 threads = 8 waves; WG tile 128(M) x 64(N); wave tile 32x32 (2x2 WMMA).
// K stepped by 32.  With ASYNC_LDS: double-buffered async global->LDS staging
// (GLOBAL_LOAD_ASYNC_TO_LDS_B128 + S_WAIT_ASYNCCNT), fragments via ds_load_b128.
#define LPAD 40   // LDS row stride (32 data + 8 pad bf16) = 80B, 16B aligned chunks

__global__ __launch_bounds__(256) void gemm_bf16(
    const unsigned short* __restrict__ A,  long strideA,
    const unsigned short* __restrict__ Bt, long strideB,
    unsigned short* __restrict__ C16, float* __restrict__ C32, long strideC,
    int K, int lda, int ldb, int ldc, float scale, int transC) {
  const int bz = blockIdx.z;
  A  += (long)bz * strideA;
  Bt += (long)bz * strideB;

  const int t    = threadIdx.x;
  const int w    = t >> 5;
  const int lane = t & 31;
  const int mBlk = blockIdx.y * 128;
  const int nBlk = blockIdx.x * 64;
  const int wm = (w >> 1) * 32;      // wave M offset in tile
  const int wn = (w & 1) * 32;       // wave N offset in tile

  v8f acc[2][2] = {};

#if ASYNC_LDS
  __shared__ unsigned short lA[2][128 * LPAD];   // 20 KB
  __shared__ unsigned short lB[2][64 * LPAD];    // 10 KB

  auto issue = [&](int buf, int k) {
    // A tile: 128 rows x 32 cols = 512 x 16B chunks; 2 per thread
    {
      int c = t;            // chunk 0..255
      int row = c >> 2, col8 = (c & 3) * 8;
      async_cp16(A + (long)(mBlk + row) * lda + k + col8,
                 &lA[buf][row * LPAD + col8]);
      c = t + 256;          // chunk 256..511
      row = c >> 2; col8 = (c & 3) * 8;
      async_cp16(A + (long)(mBlk + row) * lda + k + col8,
                 &lA[buf][row * LPAD + col8]);
    }
    // B tile: 64 rows x 32 cols = 256 x 16B chunks; 1 per thread
    {
      int row = t >> 2, col8 = (t & 3) * 8;
      async_cp16(Bt + (long)(nBlk + row) * ldb + k + col8,
                 &lB[buf][row * LPAD + col8]);
    }
  };

  issue(0, 0);
  const int nsteps = K / 32;
  for (int s = 0; s < nsteps; ++s) {
    const int buf = s & 1;
    __builtin_amdgcn_s_wait_asynccnt(0);
    __syncthreads();                      // all waves' chunks landed in LDS
    if (s + 1 < nsteps) issue(buf ^ 1, (s + 1) * 32);

    v16bf a0 = load_frag(&lA[buf][(wm + 0)  * LPAD], LPAD);
    v16bf a1 = load_frag(&lA[buf][(wm + 16) * LPAD], LPAD);
    v16bf b0 = load_frag(&lB[buf][(wn + 0)  * LPAD], LPAD);
    v16bf b1 = load_frag(&lB[buf][(wn + 16) * LPAD], LPAD);
    acc[0][0] = wmma_bf16(a0, b0, acc[0][0]);
    acc[0][1] = wmma_bf16(a0, b1, acc[0][1]);
    acc[1][0] = wmma_bf16(a1, b0, acc[1][0]);
    acc[1][1] = wmma_bf16(a1, b1, acc[1][1]);
    __syncthreads();                      // reads of buf done before overwrite
  }
#else
  const int m0d = mBlk + wm, n0d = nBlk + wn;
  for (int k = 0; k < K; k += 32) {
    const unsigned short* ap = A  + (long)m0d * lda + k;
    const unsigned short* bp = Bt + (long)n0d * ldb + k;
    if (k + 32 < K) {                     // -> global_prefetch_b8 (WGP scope)
      __builtin_prefetch(ap + 32, 0, 3);
      __builtin_prefetch(bp + 32, 0, 3);
    }
    v16bf a0 = load_frag(ap, lda);
    v16bf a1 = load_frag(ap + (long)16 * lda, lda);
    v16bf b0 = load_frag(bp, ldb);
    v16bf b1 = load_frag(bp + (long)16 * ldb, ldb);
    acc[0][0] = wmma_bf16(a0, b0, acc[0][0]);
    acc[0][1] = wmma_bf16(a0, b1, acc[0][1]);
    acc[1][0] = wmma_bf16(a1, b0, acc[1][0]);
    acc[1][1] = wmma_bf16(a1, b1, acc[1][1]);
  }
#endif

  // C/D layout: VGPR g, lanes 0-15 -> M=g, lanes 16-31 -> M=g+8; N = lane&15
  const int nl = lane & 15, hi = lane >> 4;
  for (int i = 0; i < 2; ++i)
    for (int j = 0; j < 2; ++j)
      for (int g = 0; g < 8; ++g) {
        float v = acc[i][j][g] * scale;
        long row = mBlk + wm + i * 16 + hi * 8 + g;
        long col = nBlk + wn + j * 16 + nl;
        if (transC) { long tt = row; row = col; col = tt; }
        long idx = (long)bz * strideC + row * (long)ldc + col;
        if (C32) C32[idx] = v;
        else     C16[idx] = f32_to_bf16(v);
      }
}

// ---------------- fused attention (scores -> softmax -> head-average) ----------------
// grid: (S/16, B); block: 256 (8 waves). Wave w owns key columns [w*128, w*128+128).
// Per head: QK^T via WMMA (scores live in registers), exact reference masking,
// softmax with cross-lane (shfl) + cross-wave (LDS) reductions, accumulate /H.
__global__ __launch_bounds__(256) void attn_softmax_avg(
    const unsigned short* __restrict__ Qb,   // [B,S,H*dk] bf16, pre-scaled 1/sqrt(dk)
    const unsigned short* __restrict__ Kb,   // [B,S,H*dk] bf16
    const float* __restrict__ mask,          // [B,S]
    float* __restrict__ attn_out,            // [B,S,S] f32 (output 2)
    unsigned short* __restrict__ Pb) {       // [B,S,S] bf16 (for attn @ V)
  const int b  = blockIdx.y;
  const int q0 = blockIdx.x * 16;
  const int w    = threadIdx.x >> 5;
  const int lane = threadIdx.x & 31;
  const int nl = lane & 15, hi = lane >> 4;
  const int colBase = w * 128;

  __shared__ float redMax[16][8];
  __shared__ float redSum[16][8];

  float mq[8], ms[8];
  for (int g = 0; g < 8; ++g) mq[g] = mask[(long)b * S_ + q0 + hi * 8 + g];
  for (int t = 0; t < 8; ++t) ms[t] = mask[(long)b * S_ + colBase + t * 16 + nl];

  v8f avg[8] = {};

  for (int h = 0; h < H_; ++h) {
    const unsigned short* qp = Qb + ((long)(b * S_ + q0)) * (H_ * DK_) + h * DK_;
    v16bf a0 = load_frag(qp, H_ * DK_);
    v16bf a1 = load_frag(qp + 32, H_ * DK_);

    v8f sc[8];
    for (int t = 0; t < 8; ++t) {
      const unsigned short* kp =
          Kb + ((long)(b * S_ + colBase + t * 16)) * (H_ * DK_) + h * DK_;
      v16bf b0 = load_frag(kp, H_ * DK_);
      v16bf b1 = load_frag(kp + 32, H_ * DK_);
      v8f c = {};
      c = wmma_bf16(a0, b0, c);
      c = wmma_bf16(a1, b1, c);
      sc[t] = c;
    }

    // mask + per-lane partial row max
    float pmax[8];
    for (int g = 0; g < 8; ++g) pmax[g] = -3.0e38f;
    for (int t = 0; t < 8; ++t) {
      int col = colBase + t * 16 + nl;
      for (int g = 0; g < 8; ++g) {
        int row = q0 + hi * 8 + g;
        float bad = fmaxf(col > row ? 1.f : 0.f, 1.f - mq[g] * ms[t]);
        float s = sc[t][g] - bad * 1e9f;
        sc[t][g] = s;
        pmax[g] = fmaxf(pmax[g], s);
      }
    }
    // reduce across the 16 lanes sharing the same row half
    for (int g = 0; g < 8; ++g) {
      float v = pmax[g];
      for (int off = 1; off < 16; off <<= 1) v = fmaxf(v, __shfl_xor(v, off, 32));
      pmax[g] = v;
    }
    if (nl == 0)
      for (int g = 0; g < 8; ++g) redMax[hi * 8 + g][w] = pmax[g];
    __syncthreads();
    float rmax[8];
    for (int g = 0; g < 8; ++g) {
      float m = redMax[hi * 8 + g][0];
      for (int j = 1; j < 8; ++j) m = fmaxf(m, redMax[hi * 8 + g][j]);
      rmax[g] = m;
    }

    // exp + row sum
    float psum[8] = {0.f, 0.f, 0.f, 0.f, 0.f, 0.f, 0.f, 0.f};
    for (int t = 0; t < 8; ++t)
      for (int g = 0; g < 8; ++g) {
        float e = __expf(sc[t][g] - rmax[g]);
        sc[t][g] = e;
        psum[g] += e;
      }
    for (int g = 0; g < 8; ++g) {
      float v = psum[g];
      for (int off = 1; off < 16; off <<= 1) v += __shfl_xor(v, off, 32);
      psum[g] = v;
    }
    if (nl == 0)
      for (int g = 0; g < 8; ++g) redSum[hi * 8 + g][w] = psum[g];
    __syncthreads();
    for (int g = 0; g < 8; ++g) {
      float s = 0.f;
      for (int j = 0; j < 8; ++j) s += redSum[hi * 8 + g][j];
      float inv = (1.0f / (float)H_) / s;
      for (int t = 0; t < 8; ++t) avg[t][g] += sc[t][g] * inv;
    }
    __syncthreads();   // before LDS reuse next head
  }

  for (int t = 0; t < 8; ++t)
    for (int g = 0; g < 8; ++g) {
      long row = q0 + hi * 8 + g;
      long col = colBase + t * 16 + nl;
      long idx = ((long)b * S_ + row) * S_ + col;
      float v = avg[t][g];
      attn_out[idx] = v;
      Pb[idx] = f32_to_bf16(v);
    }
}

// ---------------- launcher ----------------
extern "C" void kernel_launch(void* const* d_in, const int* in_sizes, int n_in,
                              void* d_out, int out_size, void* d_ws, size_t ws_size,
                              hipStream_t stream) {
  (void)in_sizes; (void)n_in; (void)out_size; (void)ws_size;
  const float* inputs = (const float*)d_in[0];   // [B,S,D]
  const float* mask   = (const float*)d_in[1];   // [B,S]
  const float* Wq     = (const float*)d_in[2];   // [H,D,dk]
  const float* Wk     = (const float*)d_in[3];   // [H,D,dk]
  const float* Wv     = (const float*)d_in[4];   // [D,D]
  const float* Wo     = (const float*)d_in[5];   // [D,D]

  float* out      = (float*)d_out;                       // [B,S,D]
  float* attn_out = out + (long)B_ * S_ * D_;            // [B,S,S]

  // scratch partition (bf16 stored as u16); ~104 MB total
  unsigned short* p = (unsigned short*)d_ws;
  unsigned short* Xb  = p; p += (long)B_ * S_ * D_;      // inputs bf16
  unsigned short* WqT = p; p += (long)D_ * D_;           // [H*dk, D]
  unsigned short* WkT = p; p += (long)D_ * D_;
  unsigned short* WvT = p; p += (long)D_ * D_;           // [D, D] = Wv^T
  unsigned short* WoT = p; p += (long)D_ * D_;
  unsigned short* Qb  = p; p += (long)B_ * S_ * D_;      // [B,S,H*dk] (pre-scaled)
  unsigned short* Kb  = p; p += (long)B_ * S_ * D_;
  unsigned short* Vt  = p; p += (long)B_ * D_ * S_;      // V^T per batch [B,D,S]
  unsigned short* Pb  = p; p += (long)B_ * S_ * S_;      // attn_avg bf16
  unsigned short* Hb  = p; p += (long)B_ * S_ * D_;      // Hout bf16

  const long nX = (long)B_ * S_ * D_;
  const long nW = (long)D_ * D_;
  cast_f32_to_bf16<<<(int)((nX + 255) / 256), 256, 0, stream>>>(inputs, Xb, nX);
  pack_wqk<<<(int)((nW + 255) / 256), 256, 0, stream>>>(Wq, WqT);
  pack_wqk<<<(int)((nW + 255) / 256), 256, 0, stream>>>(Wk, WkT);
  pack_wt <<<(int)((nW + 255) / 256), 256, 0, stream>>>(Wv, WvT);
  pack_wt <<<(int)((nW + 255) / 256), 256, 0, stream>>>(Wo, WoT);

  dim3 gg(D_ / 64, S_ / 128, B_);   // N/64, M/128, batches
  // Q = X @ Wq, folded 1/sqrt(dk)
  gemm_bf16<<<gg, 256, 0, stream>>>(Xb, (long)S_ * D_, WqT, 0,
                                    Qb, nullptr, (long)S_ * D_,
                                    D_, D_, D_, D_, 0.125f, 0);
  // K = X @ Wk
  gemm_bf16<<<gg, 256, 0, stream>>>(Xb, (long)S_ * D_, WkT, 0,
                                    Kb, nullptr, (long)S_ * D_,
                                    D_, D_, D_, D_, 1.0f, 0);
  // V^T = (X @ Wv)^T, stored [B, D, S] so attn@V reuses the [N,K] B-loader
  gemm_bf16<<<gg, 256, 0, stream>>>(Xb, (long)S_ * D_, WvT, 0,
                                    Vt, nullptr, (long)D_ * S_,
                                    D_, D_, D_, S_, 1.0f, 1);

  attn_softmax_avg<<<dim3(S_ / 16, B_), 256, 0, stream>>>(Qb, Kb, mask, attn_out, Pb);

  // Hout = attn_avg @ V   (Bt = V^T)
  gemm_bf16<<<gg, 256, 0, stream>>>(Pb, (long)S_ * S_, Vt, (long)D_ * S_,
                                    Hb, nullptr, (long)S_ * D_,
                                    S_, S_, S_, D_, 1.0f, 0);
  // output = Hout @ Wo  (f32 store)
  gemm_bf16<<<gg, 256, 0, stream>>>(Hb, (long)S_ * D_, WoT, 0,
                                    nullptr, out, (long)S_ * D_,
                                    D_, D_, D_, D_, 1.0f, 0);
}
